// GraphAttentionLayer_85770496901575
// MI455X (gfx1250) — matrix operand order repaired and checked by reading, not compiled, not run on hardware
//
#include <hip/hip_runtime.h>
#include <hip/hip_bf16.h>

// ---------------- types ----------------
typedef __bf16 bf16_t;
typedef bf16_t v16bf __attribute__((ext_vector_type(16)));
typedef bf16_t v8bf  __attribute__((ext_vector_type(8)));
typedef bf16_t v2bf  __attribute__((ext_vector_type(2)));
typedef float  v8f   __attribute__((ext_vector_type(8)));
typedef int    v4i   __attribute__((ext_vector_type(4)));
typedef __attribute__((address_space(1))) v4i* as1_v4i;   // global
typedef __attribute__((address_space(3))) v4i* as3_v4i;   // LDS

#define BATCH 8
#define NN    2048
#define DD    512
#define NCHUNK 64        // n-tile per attention chunk (2 PV k-steps of 32)
#define HTSTRIDE 72      // padded sHT row stride (144 B: 16B-aligned, 16-bank cycle)

#if __has_builtin(__builtin_amdgcn_global_load_async_to_lds_b128) && \
    __has_builtin(__builtin_amdgcn_s_wait_asynccnt)
#define HAVE_ASYNC_LDS 1
#endif

__device__ __forceinline__ v8f zero8() {
  v8f z;
#pragma unroll
  for (int i = 0; i < 8; ++i) z[i] = 0.0f;
  return z;
}

__device__ __forceinline__ v8f wmma_bf16(v16bf a, v16bf b, v8f c) {
  return __builtin_amdgcn_wmma_f32_16x16x32_bf16(false, a, false, b, (short)0, c, false, false);
}

union Frag { v16bf v; v8bf h[2]; };

// A fragment (16x32 bf16): lane m = lane&15, chunks at k0+8*hf and k0+16+8*hf
__device__ __forceinline__ v16bf load_a_frag(const bf16_t* rowPtr, int k0, int hf) {
  Frag f;
  f.h[0] = *(const v8bf*)(rowPtr + k0 + 8 * hf);
  f.h[1] = *(const v8bf*)(rowPtr + k0 + 16 + 8 * hf);
  return f.v;
}
// B fragment (32x16 bf16): lane col = lane&15, chunks at k0+8*hf and k0+16+8*hf
__device__ __forceinline__ v16bf load_b_frag(const bf16_t* colPtr, int k0, int hf) {
  Frag f;
  const bf16_t* p = colPtr + k0 + 8 * hf;
  f.h[0] = *(const v8bf*)(p);
  f.h[1] = *(const v8bf*)(p + 16);
  return f.v;
}

// ---------------- kernel 0: one-shot precision prep ----------------
// xb = bf16(x); Wb = bf16(W); attnT[j][i] = bf16(attn[i][j])
__global__ void k_prep(const float* __restrict__ x, const float* __restrict__ W,
                       const float* __restrict__ attn,
                       bf16_t* __restrict__ xb, bf16_t* __restrict__ Wb,
                       bf16_t* __restrict__ attnT) {
  const int XU = (BATCH * NN * DD) / 8;   // 1048576 vector units
  const int WU = (DD * DD) / 8;           // 32768
  const int AU = DD * DD;                 // 262144 scalars (transpose)
  int gid = blockIdx.x * 256 + threadIdx.x;
  if (gid < XU) {
    ((v8bf*)xb)[gid] = __builtin_convertvector(((const v8f*)x)[gid], v8bf);
  } else if (gid < XU + WU) {
    int i = gid - XU;
    ((v8bf*)Wb)[i] = __builtin_convertvector(((const v8f*)W)[i], v8bf);
  } else if (gid < XU + WU + AU) {
    int i = gid - XU - WU;
    int r = i >> 9, c = i & 511;
    attnT[(size_t)c * DD + r] = (bf16_t)attn[i];
  }
}

// ---------------- kernels 1/2: 16x64-per-wave bf16 GEMM ----------------
// wave computes a 16(m) x 64(n) tile: 4 accumulators share each A fragment.
// Fragments are double-buffered: step k+1's loads issue before step k's WMMAs
// so the in-order VMEM pipe only needs a partial loadcnt wait per step.
__device__ __forceinline__ void gemm_16x64(const bf16_t* __restrict__ A,
                                           const bf16_t* __restrict__ Brows,
                                           const float* __restrict__ bias,
                                           bf16_t* __restrict__ out,
                                           int tile) {
  const int lane = threadIdx.x & 31;
  const int hf = lane >> 4, lm = lane & 15;
  const int mt = tile >> 3;               // 1024 row tiles over B*N
  const int og = tile & 7;                // 8 col groups of 64 over DOUT
  const bf16_t* arow = A + (size_t)(mt * 16 + lm) * DD;
  const bf16_t* b0 = Brows + (size_t)(og * 64 + 0 * 16 + lm) * DD;
  const bf16_t* b1 = Brows + (size_t)(og * 64 + 1 * 16 + lm) * DD;
  const bf16_t* b2 = Brows + (size_t)(og * 64 + 2 * 16 + lm) * DD;
  const bf16_t* b3 = Brows + (size_t)(og * 64 + 3 * 16 + lm) * DD;
  v8f acc[4];
#pragma unroll
  for (int t = 0; t < 4; ++t) acc[t] = zero8();

  v16bf a  = load_a_frag(arow, 0, hf);
  v16bf f0 = load_b_frag(b0, 0, hf);
  v16bf f1 = load_b_frag(b1, 0, hf);
  v16bf f2 = load_b_frag(b2, 0, hf);
  v16bf f3 = load_b_frag(b3, 0, hf);
#pragma unroll 3
  for (int k0 = 32; k0 < DD; k0 += 32) {
    const v16bf an = load_a_frag(arow, k0, hf);
    const v16bf g0 = load_b_frag(b0, k0, hf);
    const v16bf g1 = load_b_frag(b1, k0, hf);
    const v16bf g2 = load_b_frag(b2, k0, hf);
    const v16bf g3 = load_b_frag(b3, k0, hf);
    acc[0] = wmma_bf16(a, f0, acc[0]);
    acc[1] = wmma_bf16(a, f1, acc[1]);
    acc[2] = wmma_bf16(a, f2, acc[2]);
    acc[3] = wmma_bf16(a, f3, acc[3]);
    a = an; f0 = g0; f1 = g1; f2 = g2; f3 = g3;
  }
  acc[0] = wmma_bf16(a, f0, acc[0]);
  acc[1] = wmma_bf16(a, f1, acc[1]);
  acc[2] = wmma_bf16(a, f2, acc[2]);
  acc[3] = wmma_bf16(a, f3, acc[3]);

#pragma unroll
  for (int t = 0; t < 4; ++t) {
    const float bc = bias ? bias[og * 64 + t * 16 + lm] : 0.0f;
    bf16_t* o = out + (size_t)(mt * 16 + hf * 8) * DD + og * 64 + t * 16 + lm;
#pragma unroll
    for (int r = 0; r < 8; ++r) o[(size_t)r * DD] = (bf16_t)(acc[t][r] + bc);
  }
}

__global__ void k_linear(const bf16_t* __restrict__ xb, const bf16_t* __restrict__ Wb,
                         const float* __restrict__ bias, bf16_t* __restrict__ h) {
  const int tile = blockIdx.x * (blockDim.x >> 5) + (threadIdx.x >> 5);  // 0..8191
  gemm_16x64(xb, Wb, bias, h, tile);
}

__global__ void k_ha(const bf16_t* __restrict__ h, const bf16_t* __restrict__ attnT,
                     bf16_t* __restrict__ ha) {
  const int tile = blockIdx.x * (blockDim.x >> 5) + (threadIdx.x >> 5);
  gemm_16x64(h, attnT, nullptr, ha, tile);
}

// ---------------- kernel 3: fused masked-softmax attention + ELU ----------------
// grid: B * (N/16) workgroups of 128 threads (4 waves). Online softmax over
// 64-column chunks; wave w owns score n-subtile w and output d-slice [w*128, +128).
__global__ void k_attn(const bf16_t* __restrict__ h, const bf16_t* __restrict__ ha,
                       const int* __restrict__ adj, float* __restrict__ out) {
  extern __shared__ char smem[];
  bf16_t* sHA   = (bf16_t*)smem;                        // [16][512]        16384 B
  bf16_t* sHT   = (bf16_t*)(smem + 16384);              // [512][HTSTRIDE]  73728 B
  float*  sS    = (float*)(smem + 90112);               // [16][64]         4096 B
  bf16_t* sP    = (bf16_t*)(smem + 94208);              // [16][64]         2048 B
  float*  sRed  = (float*)(smem + 96256);               // [16][8]          512 B
  float*  sM    = (float*)(smem + 96768);               // [16] pad 64B
  float*  sL    = (float*)(smem + 96832);               // [16]
  float*  sScale= (float*)(smem + 96896);               // [16]

  const int tid = threadIdx.x;            // 0..127
  const int lane = tid & 31, wave = tid >> 5;
  const int hf = lane >> 4, lm = lane & 15;
  const int b  = blockIdx.x >> 7;         // 8 batches
  const int m0 = (blockIdx.x & 127) * 16; // 128 m-tiles
  const bf16_t* hB  = h  + (size_t)b * NN * DD;
  const bf16_t* haB = ha + (size_t)b * NN * DD;
  const int*   adjB = adj + (size_t)b * NN * NN;

  // stage the query tile ha[m0:m0+16, :] into LDS (16 KB)
#ifdef HAVE_ASYNC_LDS
  for (int i = tid; i < 16 * DD / 8; i += 128)
    __builtin_amdgcn_global_load_async_to_lds_b128(
        (as1_v4i)(haB + (size_t)m0 * DD + (size_t)i * 8),
        (as3_v4i)(sHA + (size_t)i * 8), 0, 0);
  __builtin_amdgcn_s_wait_asynccnt(0);
#else
  {
    const v8bf* src = (const v8bf*)(haB + (size_t)m0 * DD);
    v8bf* dst = (v8bf*)sHA;
    for (int i = tid; i < 16 * DD / 8; i += 128) dst[i] = src[i];
  }
#endif
  if (tid < 16) { sM[tid] = -3.0e38f; sL[tid] = 0.0f; }

  v8f O[8];
#pragma unroll
  for (int t = 0; t < 8; ++t) O[t] = zero8();
  __syncthreads();

  for (int c = 0; c < NN / NCHUNK; ++c) {
    const int n0 = c * NCHUNK;

    // ---- stage transposed value chunk: sHT[d][n] = h[n0+n][d] ----
    for (int u = tid; u < (NCHUNK / 2) * (DD / 8); u += 128) {
      const int rp = u >> 6;            // row pair 0..31
      const int cg = u & 63;            // col group 0..63
      const int n = rp * 2, d = cg * 8;
      v8bf r0 = *(const v8bf*)(hB + (size_t)(n0 + n) * DD + d);
      v8bf r1 = *(const v8bf*)(hB + (size_t)(n0 + n + 1) * DD + d);
#pragma unroll
      for (int i = 0; i < 8; ++i) {
        v2bf p; p[0] = r0[i]; p[1] = r1[i];
        *(v2bf*)(sHT + (size_t)(d + i) * HTSTRIDE + n) = p;
      }
    }

    // ---- scores: wave computes S(16m x 16n) at n = n0 + wave*16, K = 512 ----
    // double-buffered fragments (A from LDS / DScnt, B from global / LOADcnt)
    v8f S = zero8();
    const bf16_t* brow = hB + (size_t)(n0 + wave * 16 + lm) * DD;  // h row as B column
    {
      v16bf sa = load_a_frag(sHA + lm * DD, 0, hf);
      v16bf sb = load_b_frag(brow, 0, hf);
#pragma unroll 3
      for (int k0 = 32; k0 < DD; k0 += 32) {
        const v16bf na = load_a_frag(sHA + lm * DD, k0, hf);
        const v16bf nb = load_b_frag(brow, k0, hf);
        S = wmma_bf16(sa, sb, S);
        sa = na; sb = nb;
      }
      S = wmma_bf16(sa, sb, S);
    }

    // prefetch next chunk's adjacency stripe (global_prefetch_b8)
    if (c + 1 < NN / NCHUNK)
      __builtin_prefetch(adjB + (size_t)(m0 + lm) * NN + n0 + NCHUNK, 0, 1);

    // ---- adjacency mask + spill scores to LDS ----
    {
      const int col = wave * 16 + lm;
#pragma unroll
      for (int r = 0; r < 8; ++r) {
        const int m = r + hf * 8;
        const int a = adjB[(size_t)(m0 + m) * NN + (n0 + col)];
        sS[m * NCHUNK + col] = (a == 0) ? -3.0e38f : S[r];
      }
    }
    __syncthreads();

    // ---- online softmax (16 rows, 8 threads/row, 8 cols each) ----
    {
      const int row = tid >> 3, seg = tid & 7;
      const float* srow = sS + row * NCHUNK + seg * 8;
      float pm = srow[0];
#pragma unroll
      for (int i = 1; i < 8; ++i) pm = fmaxf(pm, srow[i]);
      sRed[row * 8 + seg] = pm;
    }
    __syncthreads();
    if (tid < 16) {
      float cm = sRed[tid * 8];
#pragma unroll
      for (int i = 1; i < 8; ++i) cm = fmaxf(cm, sRed[tid * 8 + i]);
      const float om = sM[tid];
      const float nm = fmaxf(om, cm);
      sScale[tid] = __expf(om - nm);
      sM[tid] = nm;
    }
    __syncthreads();
    {
      const int row = tid >> 3, seg = tid & 7;
      const float nm = sM[row];
      const float* srow = sS + row * NCHUNK + seg * 8;
      bf16_t* prow = sP + row * NCHUNK + seg * 8;
      float ps = 0.0f;
#pragma unroll
      for (int i = 0; i < 8; ++i) {
        const float p = __expf(srow[i] - nm);
        ps += p;
        prow[i] = (bf16_t)p;
      }
      sRed[row * 8 + seg] = ps;
    }
    __syncthreads();
    if (tid < 16) {
      float s = 0.0f;
#pragma unroll
      for (int i = 0; i < 8; ++i) s += sRed[tid * 8 + i];
      sL[tid] = sL[tid] * sScale[tid] + s;
    }
    __syncthreads();

    // ---- rescale accumulators, then O += P @ h_chunk (d-slice per wave) ----
    {
      const v8f sc = *(const v8f*)(sScale + hf * 8);   // rows m = r + 8*hf
#pragma unroll
      for (int t = 0; t < 8; ++t)
#pragma unroll
        for (int r = 0; r < 8; ++r) O[t][r] *= sc[r];
    }
#pragma unroll
    for (int kc = 0; kc < NCHUNK / 32; ++kc) {
      const v16bf pa = load_a_frag(sP + lm * NCHUNK, kc * 32, hf);
#pragma unroll
      for (int t = 0; t < 8; ++t) {
        const int d = wave * 128 + t * 16 + lm;        // B column d -> sHT row d
        O[t] = wmma_bf16(pa, load_b_frag(sHT + (size_t)d * HTSTRIDE, kc * 32, hf), O[t]);
      }
    }
    __syncthreads();   // all LDS reads done before next chunk overwrites
  }

  // ---- epilogue: normalize by l, ELU, store f32 ----
  v8f linv = *(const v8f*)(sL + hf * 8);
#pragma unroll
  for (int r = 0; r < 8; ++r) linv[r] = 1.0f / linv[r];
#pragma unroll
  for (int t = 0; t < 8; ++t) {
    const int d = wave * 128 + t * 16 + lm;
#pragma unroll
    for (int r = 0; r < 8; ++r) {
      const int m = r + hf * 8;
      float v = O[t][r] * linv[r];
      v = (v > 0.0f) ? v : (__expf(v) - 1.0f);
      out[((size_t)b * NN + m0 + m) * DD + d] = v;
    }
  }
}

// ---------------- launcher ----------------
extern "C" void kernel_launch(void* const* d_in, const int* in_sizes, int n_in,
                              void* d_out, int out_size, void* d_ws, size_t ws_size,
                              hipStream_t stream) {
  const float* x    = (const float*)d_in[0];
  const int*   adj  = (const int*)d_in[1];
  const float* W    = (const float*)d_in[2];
  const float* bias = (const float*)d_in[3];
  const float* attn = (const float*)d_in[4];
  float* out = (float*)d_out;

  char* ws = (char*)d_ws;
  bf16_t* h_bf  = (bf16_t*)ws;                                   // 16 MB
  bf16_t* ha_bf = (bf16_t*)(ws + (size_t)16 * 1024 * 1024);      // 16 MB
  bf16_t* x_bf  = (bf16_t*)(ws + (size_t)32 * 1024 * 1024);      // 16 MB
  bf16_t* W_bf  = (bf16_t*)(ws + (size_t)48 * 1024 * 1024);      // 0.5 MB
  bf16_t* attnT = (bf16_t*)(ws + (size_t)48 * 1024 * 1024 + 524288); // 0.5 MB

  // 0) one-shot bf16 conversion of x, W and transposed attn
  const int prep_units = (BATCH * NN * DD) / 8 + (DD * DD) / 8 + DD * DD;
  k_prep<<<(prep_units + 255) / 256, 256, 0, stream>>>(x, W, attn, x_bf, W_bf, attnT);
  // 1) h = x @ W^T + b : 1024 m-tiles x 8 col-groups = 8192 wave-tiles, 8 waves/block
  k_linear<<<8192 / 8, 256, 0, stream>>>(x_bf, W_bf, bias, h_bf);
  // 2) ha = h @ attn
  k_ha<<<8192 / 8, 256, 0, stream>>>(h_bf, attnT, ha_bf);
  // 3) fused masked-softmax attention + ELU
  k_attn<<<BATCH * (NN / 16), 128, 96960, stream>>>(h_bf, ha_bf, adj, out);
}
